// MMHSA_87608742903884
// MI455X (gfx1250) — matrix operand order repaired and checked by reading, not compiled, not run on hardware
//
#include <hip/hip_runtime.h>
#include <math.h>

typedef __attribute__((ext_vector_type(16))) __bf16 v16bf;
typedef __attribute__((ext_vector_type(8)))  __bf16 v8bf;
typedef __attribute__((ext_vector_type(8)))  float  v8f;
typedef int v4i_ls __attribute__((__vector_size__(16)));   // matches builtin param type

#define AS1 __attribute__((address_space(1)))
#define AS3 __attribute__((address_space(3)))

#if defined(__has_builtin)
# if __has_builtin(__builtin_amdgcn_global_load_async_to_lds_b128)
#  define ASYNC_COPY 1
# endif
#endif
#ifndef ASYNC_COPY
# define ASYNC_COPY 0
#endif

__device__ inline __bf16 f2bf(float f) {
  unsigned u = __builtin_bit_cast(unsigned, f);
  u += 0x7FFFu + ((u >> 16) & 1u);              // round-to-nearest-even
  unsigned short h = (unsigned short)(u >> 16);
  return __builtin_bit_cast(__bf16, h);
}

#if ASYNC_COPY
__device__ inline void async_b128(const v8bf* g, v8bf* l) {
  __builtin_amdgcn_global_load_async_to_lds_b128(
      (AS1 v4i_ls*)g, (AS3 v4i_ls*)l, 0, 0);
}
#endif

__device__ inline void async_wait() {
#if ASYNC_COPY
# if __has_builtin(__builtin_amdgcn_s_wait_asynccnt)
  __builtin_amdgcn_s_wait_asynccnt(0);
# else
  asm volatile("s_wait_asynccnt 0x0" ::: "memory");
# endif
#endif
}

// ---------------- fp32 -> bf16 conversion ----------------
__global__ void cvt_f32_bf16(const float* __restrict__ in, __bf16* __restrict__ out, int n) {
  int i = blockIdx.x * blockDim.x + threadIdx.x;
  int stride = gridDim.x * blockDim.x;
  for (; i < n; i += stride) out[i] = f2bf(in[i]);
}

// ---------------- fp32 [R][C] -> bf16 [C][R] transpose ----------------
__global__ __launch_bounds__(256) void cvt_transpose(
    const float* __restrict__ in, __bf16* __restrict__ out, int R, int C)
{
  __shared__ __bf16 tile[32][33];
  int c0 = blockIdx.x * 32, r0 = blockIdx.y * 32;
  int tx = threadIdx.x, ty = threadIdx.y;   // block (32, 8)
  #pragma unroll
  for (int i = 0; i < 32; i += 8)
    tile[ty + i][tx] = f2bf(in[(size_t)(r0 + ty + i) * C + (c0 + tx)]);
  __syncthreads();
  #pragma unroll
  for (int i = 0; i < 32; i += 8)
    out[(size_t)(c0 + ty + i) * R + (r0 + tx)] = tile[tx][ty + i];
}

// ---------------- tiled NT WMMA GEMM ----------------
// C[M,N] (+)= A[M,K] * B^T, A bf16 [M][K] row-major (lda), B bf16 [N][K] row-major (ldb).
// Block tile 128x128, BK=32, double-buffered LDS. 256 threads = 8 waves;
// wave (wm 0..3, wn 0..1) owns a 32x64 region = 2x4 v_wmma 16x16x32 tiles.
// TRANSOUT: store D transposed, C[col][row] with contiguous 8-row packs.
template <typename OutT, bool ACCUM, bool TRANSOUT>
__global__ __launch_bounds__(256) void gemm_nt(
    const __bf16* __restrict__ A, int lda,
    const __bf16* __restrict__ B, int ldb,
    OutT* __restrict__ C, int ldc, int K)
{
  __shared__ __bf16 lA[2][128][32];
  __shared__ __bf16 lB[2][128][32];

  const int t  = threadIdx.x;
  const int m0 = blockIdx.y * 128;
  const int n0 = blockIdx.x * 128;
  const int lane = t & 31;
  const int wave = t >> 5;
  const int wm = wave & 3;       // 4 wave rows (32 rows each)
  const int wn = wave >> 2;      // 2 wave cols (64 cols each)
  const int r    = lane & 15;
  const int half = lane >> 4;

  // staging assignment: each thread owns 16 contiguous bf16 of one tile row
  const int lr = t >> 1;
  const int ls = (t & 1) * 16;
  const __bf16* gA = &A[(size_t)(m0 + lr) * lda + ls];
  const __bf16* gB = &B[(size_t)(n0 + lr) * ldb + ls];

  v8f acc[2][4] = {};

  auto stage = [&](int buf, int k0) {
    const v8bf* sA = (const v8bf*)(gA + k0);
    const v8bf* sB = (const v8bf*)(gB + k0);
    v8bf* dA = (v8bf*)&lA[buf][lr][ls];
    v8bf* dB = (v8bf*)&lB[buf][lr][ls];
#if ASYNC_COPY
    async_b128(sA,     dA);
    async_b128(sA + 1, dA + 1);
    async_b128(sB,     dB);
    async_b128(sB + 1, dB + 1);
#else
    dA[0] = sA[0]; dA[1] = sA[1];
    dB[0] = sB[0]; dB[1] = sB[1];
#endif
  };

  auto compute = [&](int buf) {
    union { v16bf v; v8bf h[2]; } af[2];
    #pragma unroll
    for (int mt = 0; mt < 2; ++mt) {
      const __bf16* ap = &lA[buf][wm * 32 + mt * 16 + r][0];
      af[mt].h[0] = *(const v8bf*)&ap[half * 16];       // K = 16h .. 16h+7
      af[mt].h[1] = *(const v8bf*)&ap[half * 16 + 8];   // K = 16h+8 .. 16h+15
    }
    v16bf bfr[4];
    #pragma unroll
    for (int nt = 0; nt < 4; ++nt)
      bfr[nt] = *(const v16bf*)&lB[buf][wn * 64 + nt * 16 + r][half * 16];
    #pragma unroll
    for (int mt = 0; mt < 2; ++mt)
      #pragma unroll
      for (int nt = 0; nt < 4; ++nt)
        acc[mt][nt] = __builtin_amdgcn_wmma_f32_16x16x32_bf16(
            false, af[mt].v, false, bfr[nt], (short)0, acc[mt][nt], false, false);
  };

  // software pipeline: stage(i+1) overlaps compute(i); one barrier per K-step
  stage(0, 0);
  async_wait();
  __syncthreads();
  int cur = 0;
  for (int k0 = 32; k0 < K; k0 += 32) {
    stage(cur ^ 1, k0);
    compute(cur);
    async_wait();
    __syncthreads();
    cur ^= 1;
  }
  compute(cur);

  // epilogue: C/D fragment layout row = 8*half + vgpr, col = lane&15
  #pragma unroll
  for (int mt = 0; mt < 2; ++mt) {
    #pragma unroll
    for (int nt = 0; nt < 4; ++nt) {
      int row = m0 + wm * 32 + mt * 16 + half * 8;
      int col = n0 + wn * 64 + nt * 16 + r;
      if constexpr (TRANSOUT) {
        v8bf pk;
        #pragma unroll
        for (int v = 0; v < 8; ++v) pk[v] = f2bf(acc[mt][nt][v]);
        *(v8bf*)&C[(size_t)col * ldc + row] = pk;   // contiguous 16B store
      } else {
        #pragma unroll
        for (int v = 0; v < 8; ++v) {
          float val = acc[mt][nt][v];
          size_t idx = (size_t)(row + v) * ldc + col;
          if constexpr (ACCUM) val += (float)C[idx];
          if constexpr (__is_same(OutT, float)) C[idx] = val;
          else                                  C[idx] = f2bf(val);
        }
      }
    }
  }
}

// ---------------- row softmax with causal mask + scale ----------------
__global__ __launch_bounds__(256) void softmax_rows(
    const float* __restrict__ sc, __bf16* __restrict__ attn, int S, float scale)
{
  const int row = blockIdx.x;
  const float* rp = sc + (size_t)row * S;
  __bf16* op = attn + (size_t)row * S;
  __shared__ float red[256];
  const int t = threadIdx.x;

  float mx = -3.4e38f;
  for (int c = t; c < S; c += 256) {
    float v = (c <= row) ? rp[c] * scale : -1e9f;
    mx = fmaxf(mx, v);
  }
  red[t] = mx; __syncthreads();
  for (int s = 128; s > 0; s >>= 1) { if (t < s) red[t] = fmaxf(red[t], red[t + s]); __syncthreads(); }
  mx = red[0]; __syncthreads();

  float sum = 0.f;
  for (int c = t; c < S; c += 256)
    if (c <= row) sum += __expf(rp[c] * scale - mx);
  red[t] = sum; __syncthreads();
  for (int s = 128; s > 0; s >>= 1) { if (t < s) red[t] += red[t + s]; __syncthreads(); }
  float inv = 1.f / red[0];

  for (int c = t; c < S; c += 256) {
    float v = (c <= row) ? __expf(rp[c] * scale - mx) * inv : 0.f;
    op[c] = f2bf(v);
  }
}

// ---------------- launch ----------------
extern "C" void kernel_launch(void* const* d_in, const int* in_sizes, int n_in,
                              void* d_out, int out_size, void* d_ws, size_t ws_size,
                              hipStream_t stream)
{
  (void)in_sizes; (void)n_in; (void)out_size; (void)ws_size;
  const float* x  = (const float*)d_in[0];
  // d_in[1] = causal tril mask: applied analytically (col <= row) in softmax
  const float* Wq = (const float*)d_in[2];
  const float* Wk = (const float*)d_in[3];
  const float* Wv = (const float*)d_in[4];
  const float* Wo = (const float*)d_in[5];
  float* out = (float*)d_out;

  constexpr int Bb = 4, S = 2048, D = 2048;
  constexpr int BS = Bb * S;

  char* ws = (char*)d_ws;
  size_t off = 0;
  auto take = [&](size_t bytes) {
    void* p = ws + off;
    off = (off + bytes + 255) & ~(size_t)255;
    return p;
  };
  __bf16* xb  = (__bf16*)take((size_t)BS * D * 2);   // x bf16, reused as O later
  __bf16* Wqt = (__bf16*)take((size_t)D * D * 2);    // W^T bf16
  __bf16* Wkt = (__bf16*)take((size_t)D * D * 2);
  __bf16* Wvt = (__bf16*)take((size_t)D * D * 2);
  __bf16* Wot = (__bf16*)take((size_t)D * D * 2);
  __bf16* Qb  = (__bf16*)take((size_t)BS * D * 2);
  __bf16* Kb  = (__bf16*)take((size_t)BS * D * 2);
  __bf16* Vt  = (__bf16*)take((size_t)BS * D * 2);   // V transposed per batch: [D][S]
  float*  sc  = (float*) take((size_t)S * S * 4);
  __bf16* at  = (__bf16*)take((size_t)S * S * 2);
  __bf16* Ob  = xb;   // x is dead after QKV projections

  cvt_f32_bf16<<<4096, 256, 0, stream>>>(x, xb, BS * D);
  dim3 tblk(32, 8), tgrd(D / 32, D / 32);
  cvt_transpose<<<tgrd, tblk, 0, stream>>>(Wq, Wqt, D, D);
  cvt_transpose<<<tgrd, tblk, 0, stream>>>(Wk, Wkt, D, D);
  cvt_transpose<<<tgrd, tblk, 0, stream>>>(Wv, Wvt, D, D);
  cvt_transpose<<<tgrd, tblk, 0, stream>>>(Wo, Wot, D, D);

  dim3 blk(256);
  dim3 gProj(D / 128, BS / 128);
  gemm_nt<__bf16, false, false><<<gProj, blk, 0, stream>>>(xb, D, Wqt, D, Qb, D, D);
  gemm_nt<__bf16, false, false><<<gProj, blk, 0, stream>>>(xb, D, Wkt, D, Kb, D, D);
  // V projection written transposed, per batch: Vt_b[d][s]
  dim3 gVb(D / 128, S / 128);
  for (int b = 0; b < Bb; ++b)
    gemm_nt<__bf16, false, true><<<gVb, blk, 0, stream>>>(
        xb + (size_t)b * S * D, D, Wvt, D, Vt + (size_t)b * D * S, S, D);

  dim3 gScore(S / 128, S / 128);
  dim3 gAV(512 / 128, S / 128);
  for (int b = 0; b < Bb; ++b) {
    const __bf16* Qp  = Qb + (size_t)b * S * D;
    const __bf16* Kp  = Kb + (size_t)b * S * D;
    const __bf16* Vtp = Vt + (size_t)b * D * S;
    __bf16*       Op  = Ob + (size_t)b * S * D;
    for (int l = 0; l < 4; ++l) {
      int ks = l * 512;  // qk and head slice offsets coincide (512-wide slices)
      if (l == 0)
        gemm_nt<float, false, false><<<gScore, blk, 0, stream>>>(Qp + ks, D, Kp + ks, D, sc, S, 512);
      else
        gemm_nt<float, true,  false><<<gScore, blk, 0, stream>>>(Qp + ks, D, Kp + ks, D, sc, S, 512);
      float scale = 1.0f / sqrtf(512.0f * (float)(l + 1));
      softmax_rows<<<S, 256, 0, stream>>>(sc, at, S, scale);
      // out slice = attn @ V[:, ks:ks+512] == attn @ (Vt rows ks..ks+511)^T
      gemm_nt<__bf16, false, false><<<gAV, blk, 0, stream>>>(
          at, S, Vtp + (size_t)ks * S, S, Op + ks, D, S);
    }
  }

  gemm_nt<float, false, false><<<gProj, blk, 0, stream>>>(Ob, D, Wot, D, out, D, D);
}
